// Multiplicative_29600914604039
// MI455X (gfx1250) — matrix operand order repaired and checked by reading, not compiled, not run on hardware
//
#include <hip/hip_runtime.h>
#include <hip/hip_bf16.h>
#include <math.h>
#include <stdint.h>

// ---------------------------------------------------------------------------
// Bilinear attention on MI455X (gfx1250), wave32, bf16 WMMA + async-LDS path
// with double-buffered (overlapped) global->LDS DMA.
//   B=8, Q=2048, K=2048, D=512
//   out = [context (B*Q*D f32), dist (B*Q*K f32)]
// ---------------------------------------------------------------------------

#define BB 8
#define QQ 2048
#define KK 2048
#define DD 512

typedef __attribute__((ext_vector_type(16))) __bf16 v16bf;
typedef __attribute__((ext_vector_type(8)))  __bf16 v8bf;
typedef __attribute__((ext_vector_type(8)))  float  v8f;

static __device__ __forceinline__ v8f wmma_bf16(v16bf a, v16bf b, v8f c) {
    // (neg_a, A, neg_b, B, c_mod, C, reuse_a, reuse_b)
    return __builtin_amdgcn_wmma_f32_16x16x32_bf16(false, a, false, b, (short)0, c,
                                                   false, false);
}

// ---------------------------------------------------------------------------
// Elementwise f32 -> bf16 convert (grid-stride).
// ---------------------------------------------------------------------------
__global__ __launch_bounds__(256) void cvt_flat(const float* __restrict__ in,
                                                __bf16* __restrict__ out, size_t n) {
    size_t i = (size_t)blockIdx.x * blockDim.x + threadIdx.x;
    size_t stride = (size_t)gridDim.x * blockDim.x;
    for (; i < n; i += stride) out[i] = (__bf16)in[i];
}

// ---------------------------------------------------------------------------
// Batched transpose + convert: out[b][c][r] = (bf16) in[b][r][c].
// block (32,8), grid (C/32, R/32, B). LDS 32x33 f32 tile (pad kills conflicts).
// ---------------------------------------------------------------------------
__global__ __launch_bounds__(256) void transpose_cvt(const float* __restrict__ in,
                                                     __bf16* __restrict__ out,
                                                     int R, int C) {
    __shared__ float tile[32][33];
    const int b = blockIdx.z;
    in  += (size_t)b * R * C;
    out += (size_t)b * R * C;
    const int c0 = blockIdx.x * 32, r0 = blockIdx.y * 32;
    const int tx = threadIdx.x, ty = threadIdx.y;
    #pragma unroll
    for (int j = 0; j < 4; ++j)
        tile[ty + 8 * j][tx] = in[(size_t)(r0 + ty + 8 * j) * C + c0 + tx];
    __syncthreads();
    #pragma unroll
    for (int j = 0; j < 4; ++j)
        out[(size_t)(c0 + ty + 8 * j) * R + r0 + tx] = (__bf16)tile[tx][ty + 8 * j];
}

// ---------------------------------------------------------------------------
// Shared WMMA GEMM core. Macro-tile 128(M) x 64(N), 256 threads = 8 wave32s,
// each wave owns a 32x32 sub-tile -> 2 A-frags x 2 B-frags -> 4 WMMA / K-step.
//
//  A (bf16, row-major [M][lda]) is staged global->LDS with
//  GLOBAL_LOAD_ASYNC_TO_LDS_B128 (ASYNCcnt), DOUBLE-BUFFERED: tile i+1's DMA
//  overlaps tile i's WMMAs; s_wait_asynccnt 2 retires tile i (async loads
//  complete in order) while tile i+1's two loads stay in flight.
//  WMMA A-fragment layout:
//    lane<16 : M=lane,    K in {0..15};  lane>=16: K in {16..31} (contiguous, as B)
//  B (bf16, "N-major" [N][ldb], contiguous in K) is read DIRECTLY from global
//  (192MB L2-backed):  lane L -> N = L%16, K = 16*(L/16)+e == one 32B vector.
// ---------------------------------------------------------------------------
struct WaveCoord { int wr, wc, half, lm; };

static __device__ __forceinline__ WaveCoord wave_coord() {
    const int tid = threadIdx.x;
    const int wave = tid >> 5, lane = tid & 31;
    WaveCoord w;
    w.wr = (wave & 3) * 32;     // wave row offset within 128
    w.wc = (wave >> 2) * 32;    // wave col offset within 64
    w.half = lane >> 4;
    w.lm = lane & 15;
    return w;
}

// One thread's share of a 128x32 bf16 A tile: 32B = two b128 async loads.
static __device__ __forceinline__ void async_copy_tile(const __bf16* g, uint32_t lds) {
    asm volatile("global_load_async_to_lds_b128 %0, %1, off"
                 :: "v"(lds), "v"(g) : "memory");
    asm volatile("global_load_async_to_lds_b128 %0, %1, off"
                 :: "v"(lds + 16u), "v"(g + 8) : "memory");
}

static __device__ __forceinline__ void gemm_core(
        const __bf16* __restrict__ Abase, int lda,
        const __bf16* __restrict__ Bbase, int ldb,
        int kExtent, int M0, int N0,
        __bf16* AsLds /* 2 x 128*32 */, v8f acc[2][2]) {
    const int tid = threadIdx.x;
    const WaveCoord w = wave_coord();

    const int cr = tid >> 1;             // copy row 0..127
    const int cc = (tid & 1) * 16;       // copy col 0 / 16
    const uint32_t ldsA =
        (uint32_t)(uintptr_t)AsLds + (uint32_t)(cr * 32 + cc) * 2u;
    const __bf16* gA = Abase + (size_t)(M0 + cr) * lda + cc;

    const int nIter = kExtent >> 5;
    async_copy_tile(gA, ldsA);                       // tile 0 -> buffer 0

    for (int i = 0; i < nIter; ++i) {
        const int k0 = i << 5;
        // WAR: buffer (i+1)&1 was read by all waves during iteration i-1.
        __syncthreads();
        if (i + 1 < nIter) {
            async_copy_tile(gA + k0 + 32, ldsA + (uint32_t)((i + 1) & 1) * 8192u);
            __builtin_prefetch(Bbase + (size_t)(N0 + w.wc + w.lm) * ldb + k0 + 32, 0, 3);
            // retire tile i's 2 loads; tile i+1's 2 stay in flight (in-order)
            asm volatile("s_wait_asynccnt 2" ::: "memory");
        } else {
            asm volatile("s_wait_asynccnt 0" ::: "memory");
        }
        __syncthreads();                 // tile i visible to all waves

        const __bf16* buf = AsLds + (i & 1) * (128 * 32);

        // ---- A fragments (rows wr, wr+16) from LDS ----
        v16bf afr[2];
        #pragma unroll
        for (int ii = 0; ii < 2; ++ii) {
            const __bf16* row = buf + (w.wr + ii * 16 + w.lm) * 32 + w.half * 16;
            v8bf lo = *(const v8bf*)(row);        // K = half*16 .. +7
            v8bf hi = *(const v8bf*)(row + 8);    // K = half*16+8 .. +15
            afr[ii] = __builtin_shufflevector(lo, hi,
                        0,1,2,3,4,5,6,7,8,9,10,11,12,13,14,15);
        }
        // ---- B fragments (cols wc, wc+16) straight from global ----
        v16bf bfr[2];
        #pragma unroll
        for (int j = 0; j < 2; ++j) {
            const __bf16* bp = Bbase +
                (size_t)(N0 + w.wc + j * 16 + w.lm) * ldb + k0 + w.half * 16;
            bfr[j] = *(const v16bf*)bp;           // 32B contiguous, 32B aligned
        }
        #pragma unroll
        for (int ii = 0; ii < 2; ++ii)
            #pragma unroll
            for (int j = 0; j < 2; ++j)
                acc[ii][j] = wmma_bf16(afr[ii], bfr[j], acc[ii][j]);
    }
}

// ---------------------------------------------------------------------------
// GEMM 1: qWb[m][e] = sum_d qb[m][d] * Wt[e][d]   (M=B*Q, N=D, K=D), bf16 out
// ---------------------------------------------------------------------------
__global__ __launch_bounds__(256) void gemm_qw(const __bf16* __restrict__ qb,
                                               const __bf16* __restrict__ Wt,
                                               __bf16* __restrict__ qWb) {
    __shared__ __align__(16) __bf16 As[2 * 128 * 32];
    const int M0 = blockIdx.x * 128, N0 = blockIdx.y * 64;
    v8f acc[2][2] = {};
    gemm_core(qb, DD, Wt, DD, DD, M0, N0, As, acc);

    const WaveCoord w = wave_coord();
    #pragma unroll
    for (int i = 0; i < 2; ++i) {
        const int gm0 = M0 + w.wr + i * 16 + w.half * 8;
        #pragma unroll
        for (int j = 0; j < 2; ++j) {
            const int gn = N0 + w.wc + j * 16 + w.lm;
            #pragma unroll
            for (int v = 0; v < 8; ++v)
                qWb[(size_t)(gm0 + v) * DD + gn] = (__bf16)acc[i][j][v];
        }
    }
}

// ---------------------------------------------------------------------------
// GEMM 2: scores[b][q][kk] = sum_d qWb[b][q][d]*kb[b][kk][d]; mask+scale fused
// ---------------------------------------------------------------------------
__global__ __launch_bounds__(256) void gemm_scores(const __bf16* __restrict__ qWb,
                                                   const __bf16* __restrict__ kb,
                                                   const int* __restrict__ kmask,
                                                   float* __restrict__ dist,
                                                   float scale) {
    __shared__ __align__(16) __bf16 As[2 * 128 * 32];
    const int b = blockIdx.z;
    const int M0 = blockIdx.x * 128, N0 = blockIdx.y * 64;
    v8f acc[2][2] = {};
    gemm_core(qWb + (size_t)b * QQ * DD, DD,
              kb  + (size_t)b * KK * DD, DD,
              DD, M0, N0, As, acc);

    const WaveCoord w = wave_coord();
    const float neg = -1e20f * scale;
    float* dbase = dist + (size_t)b * QQ * KK;
    #pragma unroll
    for (int j = 0; j < 2; ++j) {
        const int gn = N0 + w.wc + j * 16 + w.lm;
        const bool live = kmask[(size_t)b * KK + gn] != 0;
        #pragma unroll
        for (int i = 0; i < 2; ++i) {
            const int gm0 = M0 + w.wr + i * 16 + w.half * 8;
            #pragma unroll
            for (int v = 0; v < 8; ++v)
                dbase[(size_t)(gm0 + v) * KK + gn] = live ? acc[i][j][v] * scale : neg;
        }
    }
}

// ---------------------------------------------------------------------------
// Softmax over last axis; writes fp32 dist in place + bf16 copy for GEMM 3.
// ---------------------------------------------------------------------------
__global__ __launch_bounds__(256) void softmax_rows(float* __restrict__ dist,
                                                    __bf16* __restrict__ distb) {
    __shared__ float red[256];
    float*  p  = dist  + (size_t)blockIdx.x * KK;
    __bf16* pb = distb + (size_t)blockIdx.x * KK;
    const int t = threadIdx.x;

    float v[8];
    float m = -INFINITY;
    #pragma unroll
    for (int i = 0; i < 8; ++i) { v[i] = p[t + i * 256]; m = fmaxf(m, v[i]); }
    red[t] = m; __syncthreads();
    for (int s = 128; s > 0; s >>= 1) {
        if (t < s) red[t] = fmaxf(red[t], red[t + s]);
        __syncthreads();
    }
    const float rowmax = red[0];
    __syncthreads();

    float sum = 0.0f;
    #pragma unroll
    for (int i = 0; i < 8; ++i) { v[i] = __expf(v[i] - rowmax); sum += v[i]; }
    red[t] = sum; __syncthreads();
    for (int s = 128; s > 0; s >>= 1) {
        if (t < s) red[t] += red[t + s];
        __syncthreads();
    }
    const float inv = 1.0f / red[0];
    #pragma unroll
    for (int i = 0; i < 8; ++i) {
        const float d = v[i] * inv;
        p[t + i * 256]  = d;
        pb[t + i * 256] = (__bf16)d;
    }
}

// ---------------------------------------------------------------------------
// GEMM 3: context[b][q][d] = sum_kk distb[b][q][kk] * vbT[b][d][kk]
// ---------------------------------------------------------------------------
__global__ __launch_bounds__(256) void gemm_ctx(const __bf16* __restrict__ distb,
                                                const __bf16* __restrict__ vbT,
                                                float* __restrict__ ctx) {
    __shared__ __align__(16) __bf16 As[2 * 128 * 32];
    const int b = blockIdx.z;
    const int M0 = blockIdx.x * 128, N0 = blockIdx.y * 64;
    v8f acc[2][2] = {};
    gemm_core(distb + (size_t)b * QQ * KK, KK,
              vbT   + (size_t)b * DD * KK, KK,
              KK, M0, N0, As, acc);

    const WaveCoord w = wave_coord();
    float* cbase = ctx + (size_t)b * QQ * DD;
    #pragma unroll
    for (int i = 0; i < 2; ++i) {
        const int gm0 = M0 + w.wr + i * 16 + w.half * 8;
        #pragma unroll
        for (int j = 0; j < 2; ++j) {
            const int gn = N0 + w.wc + j * 16 + w.lm;
            #pragma unroll
            for (int v = 0; v < 8; ++v)
                cbase[(size_t)(gm0 + v) * DD + gn] = acc[i][j][v];
        }
    }
}

// ---------------------------------------------------------------------------
extern "C" void kernel_launch(void* const* d_in, const int* in_sizes, int n_in,
                              void* d_out, int out_size, void* d_ws, size_t ws_size,
                              hipStream_t stream) {
    (void)in_sizes; (void)n_in; (void)out_size; (void)ws_size;
    const float* query = (const float*)d_in[0];   // [B,Q,D]
    const float* keyf  = (const float*)d_in[1];   // [B,K,D]
    const float* value = (const float*)d_in[2];   // [B,K,D]
    const int*   kmask = (const int*)d_in[3];     // [B,K]
    const float* wmat  = (const float*)d_in[4];   // [D,D]

    float* ctx  = (float*)d_out;                        // B*Q*D
    float* dist = (float*)d_out + (size_t)BB * QQ * DD; // B*Q*K

    // workspace layout (bf16 elements), ~135 MB total
    __bf16* ws    = (__bf16*)d_ws;
    __bf16* qb    = ws;                                   // B*Q*D
    __bf16* kb    = qb  + (size_t)BB * QQ * DD;           // B*K*D
    __bf16* vbT   = kb  + (size_t)BB * KK * DD;           // B*D*K (transposed)
    __bf16* Wt    = vbT + (size_t)BB * DD * KK;           // D*D   (transposed)
    __bf16* qWb   = Wt  + (size_t)DD * DD;                // B*Q*D
    __bf16* distb = qWb + (size_t)BB * QQ * DD;           // B*Q*K

    const float scale = 0.044194173824159216f;            // 1/sqrt(512)

    dim3 blk(256);
    // precision staging: f32 -> bf16 (+ transposes so all GEMM reads are
    // contiguous and match the WMMA fragment layouts exactly)
    cvt_flat<<<dim3(4096), blk, 0, stream>>>(query, qb, (size_t)BB * QQ * DD);
    cvt_flat<<<dim3(4096), blk, 0, stream>>>(keyf,  kb, (size_t)BB * KK * DD);
    transpose_cvt<<<dim3(DD / 32, DD / 32, 1), dim3(32, 8), 0, stream>>>(wmat,  Wt,  DD, DD);
    transpose_cvt<<<dim3(DD / 32, KK / 32, BB), dim3(32, 8), 0, stream>>>(value, vbT, KK, DD);

    gemm_qw    <<<dim3((BB * QQ) / 128, DD / 64),  blk, 0, stream>>>(qb, Wt, qWb);
    gemm_scores<<<dim3(QQ / 128, KK / 64, BB),     blk, 0, stream>>>(qWb, kb, kmask, dist, scale);
    softmax_rows<<<dim3(BB * QQ),                  blk, 0, stream>>>(dist, distb);
    gemm_ctx   <<<dim3(QQ / 128, DD / 64, BB),     blk, 0, stream>>>(distb, vbT, ctx);
}
